// LiquidNeuralNetwork_10746008174614
// MI455X (gfx1250) — matrix operand order repaired and checked by reading, not compiled, not run on hardware
//
#include <hip/hip_runtime.h>
#include <hip/hip_bf16.h>

// ---------------------------------------------------------------------------
// Liquid NN on gfx1250 (MI455X), wave32 + WMMA bf16 split-precision.
//
//   xin_t + h@W_h  ==  [x_t | h] @ [W_in ; W_h]   (concatenated K = 1536)
//
// 8 workgroups x 512 threads; each WG owns 16 batch rows and runs the full
// 512-step scan with only workgroup barriers. A-matrix (x_t | h, split bf16
// hi/lo) lives in LDS; weights are pre-packed into WMMA-fragment order in
// d_ws so each lane's B fragment is a contiguous 32B global load.
// Each 16x16 product tile uses 3x V_WMMA_F32_16X16X32_BF16 (hi*hi + hi*lo +
// lo*hi) for ~fp32 accuracy with f32 accumulation.
// ---------------------------------------------------------------------------

#define INPUT  512
#define HIDDEN 1024
#define OUTPUT 256
#define BATCH  128
#define SEQ    512

#define KCAT (INPUT + HIDDEN)     // 1536 concatenated K
#define NKT  (KCAT / 32)          // 48 k-tiles (recurrent GEMM)
#define NNT  (HIDDEN / 16)        // 64 n-tiles
#define NKT2 (HIDDEN / 32)        // 32 k-tiles (output GEMM)
#define NNT2 (OUTPUT / 16)        // 16 n-tiles (output GEMM)
#define AP   (KCAT + 8)           // padded LDS row (elements) -> no bank conflicts

typedef __attribute__((ext_vector_type(16))) __bf16          v16bf;
typedef __attribute__((ext_vector_type(8)))  __bf16          v8bf;
typedef __attribute__((ext_vector_type(8)))  float           v8f;
typedef __attribute__((ext_vector_type(4)))  float           f32x4;
typedef __attribute__((ext_vector_type(4)))  unsigned short  u16x4;

__device__ __forceinline__ unsigned short bf16_rne(float f) {
  unsigned int u = __float_as_uint(f);
  u += 0x7fffu + ((u >> 16) & 1u);
  return (unsigned short)(u >> 16);
}
__device__ __forceinline__ float bf16_f(unsigned short h) {
  return __uint_as_float(((unsigned int)h) << 16);
}

// ---- prep: pack [W_in ; W_h] (1536x1024) into fragment-major split bf16 ----
// packed index p = ((kt*64 + nt)*32 + lane)*16 + j
//   K = kt*32 + (j<8 ? 0 : 16) + (lane<16 ? 0 : 8) + (j&7) ,  N = nt*16 + (lane&15)
__global__ void pack_wcat(const float* __restrict__ W_in,
                          const float* __restrict__ W_h,
                          unsigned short* __restrict__ Bhi,
                          unsigned short* __restrict__ Blo) {
  int p = blockIdx.x * blockDim.x + threadIdx.x;
  if (p >= KCAT * HIDDEN) return;
  int j  = p & 15;
  int l  = (p >> 4) & 31;
  int nt = (p >> 9) & 63;
  int kt = p >> 15;
  int K = kt * 32 + ((j & 8) << 1) + ((l < 16) ? 0 : 8) + (j & 7);
  int N = nt * 16 + (l & 15);
  float v = (K < INPUT) ? W_in[K * HIDDEN + N] : W_h[(K - INPUT) * HIDDEN + N];
  unsigned short hi = bf16_rne(v);
  Bhi[p] = hi;
  Blo[p] = bf16_rne(v - bf16_f(hi));
}

// ---- prep: pack W_out (1024x256) the same way -----------------------------
__global__ void pack_wout(const float* __restrict__ W_out,
                          unsigned short* __restrict__ BOhi,
                          unsigned short* __restrict__ BOlo) {
  int p = blockIdx.x * blockDim.x + threadIdx.x;
  if (p >= HIDDEN * OUTPUT) return;
  int j  = p & 15;
  int l  = (p >> 4) & 31;
  int nt = (p >> 9) & 15;
  int kt = p >> 13;
  int K = kt * 32 + ((j & 8) << 1) + ((l < 16) ? 0 : 8) + (j & 7);
  int N = nt * 16 + (l & 15);
  float v = W_out[K * OUTPUT + N];
  unsigned short hi = bf16_rne(v);
  BOhi[p] = hi;
  BOlo[p] = bf16_rne(v - bf16_f(hi));
}

// ---- main: persistent recurrent kernel, one WG = 16 batch rows ------------
__global__ __launch_bounds__(512)
void lnn_scan(const float* __restrict__ x,
              const float* __restrict__ b_in,
              const float* __restrict__ b_h,
              const float* __restrict__ tau,
              const float* __restrict__ b_out,
              const unsigned short* __restrict__ Bhi,
              const unsigned short* __restrict__ Blo,
              const unsigned short* __restrict__ BOhi,
              const unsigned short* __restrict__ BOlo,
              float* __restrict__ out) {
  extern __shared__ unsigned short smem[];
  unsigned short* Ahi = smem;                    // 16 * AP  (x_t | h) hi
  unsigned short* Alo = smem + 16 * AP;          // 16 * AP  (x_t | h) lo
  float* bc   = (float*)(smem + 32 * AP);        // b_in + b_h     [HIDDEN]
  float* itau = bc + HIDDEN;                     // 1 / tau        [HIDDEN]

  const int tid  = threadIdx.x;
  const int lane = tid & 31;
  const int w    = tid >> 5;                     // wave 0..15
  const int b0   = blockIdx.x * 16;

  // one-time init: fused bias, inverse tau, h = 0
  for (int i = tid; i < HIDDEN; i += 512) {
    bc[i]   = b_in[i] + b_h[i];
    itau[i] = 1.0f / tau[i];
  }
  for (int i = tid; i < 16 * HIDDEN; i += 512) {
    int mm = i >> 10, kk = i & (HIDDEN - 1);
    Ahi[mm * AP + INPUT + kk] = 0;
    Alo[mm * AP + INPUT + kk] = 0;
  }

  const int m    = lane & 15;          // A row / C-matrix base
  const int half = (lane < 16) ? 0 : 8;
  const int arow = m * AP;

  v8f C[4];

#pragma unroll 1
  for (int t = 0; t < SEQ; ++t) {
    // ---- stage x[:, t, :] as split bf16 into A[:, 0:INPUT] (vectorized) ----
    // 16 rows x 512 cols = 2048 float4 groups; 4 groups per thread per step.
#pragma unroll
    for (int g = 0; g < 4; ++g) {
      int idx4 = tid + g * 512;                 // 0..2047
      int mm   = idx4 >> 7;                     // 128 float4 per row
      int ii   = (idx4 & 127) << 2;             // column (elements)
      f32x4 xv = *(const f32x4*)(x + (((long)(b0 + mm)) * SEQ + t) * INPUT + ii);
      u16x4 hv, lv;
#pragma unroll
      for (int c = 0; c < 4; ++c) {
        unsigned short hb = bf16_rne(xv[c]);
        hv[c] = hb;
        lv[c] = bf16_rne(xv[c] - bf16_f(hb));
      }
      *(u16x4*)(Ahi + mm * AP + ii) = hv;       // ds_store_b64
      *(u16x4*)(Alo + mm * AP + ii) = lv;       // ds_store_b64
    }
    __syncthreads();   // x staged + all h updates from step t-1 visible

    C[0] = (v8f){0,0,0,0,0,0,0,0};
    C[1] = (v8f){0,0,0,0,0,0,0,0};
    C[2] = (v8f){0,0,0,0,0,0,0,0};
    C[3] = (v8f){0,0,0,0,0,0,0,0};

    // ---- pre = [x_t | h] @ [W_in ; W_h]  (split-bf16, f32 accumulate) ----
    for (int kt = 0; kt < NKT; ++kt) {
      const int kbase = kt * 32;
      v8bf a0 = *(const v8bf*)(Ahi + arow + kbase + half);
      v8bf a1 = *(const v8bf*)(Ahi + arow + kbase + 16 + half);
      v8bf l0 = *(const v8bf*)(Alo + arow + kbase + half);
      v8bf l1 = *(const v8bf*)(Alo + arow + kbase + 16 + half);
      v16bf ahi = __builtin_shufflevector(a0, a1, 0,1,2,3,4,5,6,7,8,9,10,11,12,13,14,15);
      v16bf alo = __builtin_shufflevector(l0, l1, 0,1,2,3,4,5,6,7,8,9,10,11,12,13,14,15);
#pragma unroll
      for (int q = 0; q < 4; ++q) {
        const int  nt = w * 4 + q;
        const long fb = (((long)kt * NNT + nt) * 32 + lane) * 16;
        v16bf bhi = *(const v16bf*)(Bhi + fb);
        v16bf blo = *(const v16bf*)(Blo + fb);
        C[q] = __builtin_amdgcn_wmma_f32_16x16x32_bf16(false, ahi, false, bhi, (short)0, C[q], false, false);
        C[q] = __builtin_amdgcn_wmma_f32_16x16x32_bf16(false, ahi, false, blo, (short)0, C[q], false, false);
        C[q] = __builtin_amdgcn_wmma_f32_16x16x32_bf16(false, alo, false, bhi, (short)0, C[q], false, false);
      }
    }
    __syncthreads();   // all waves done reading A before h is rewritten

    // ---- epilogue: dx = tanh(pre + bias);  h += (dx - h) * (1/tau) ----
#pragma unroll
    for (int q = 0; q < 4; ++q) {
      const int n = (w * 4 + q) * 16 + (lane & 15);
#pragma unroll
      for (int e = 0; e < 8; ++e) {
        const int mr   = e + half;
        const int hidx = mr * AP + INPUT + n;
        float hold = bf16_f(Ahi[hidx]) + bf16_f(Alo[hidx]);
        float dx   = tanhf(C[q][e] + bc[n]);
        float hnew = hold + (dx - hold) * itau[n];
        unsigned short hh = bf16_rne(hnew);
        Ahi[hidx] = hh;
        Alo[hidx] = bf16_rne(hnew - bf16_f(hh));
      }
    }
    // no barrier needed here: next staging writes only the disjoint x region,
    // and the post-staging barrier orders these h writes before next reads.
  }

  __syncthreads();

  // ---- out = h_final @ W_out + b_out  (one n-tile per wave) ----
  v8f Co = (v8f){0,0,0,0,0,0,0,0};
  for (int kt = 0; kt < NKT2; ++kt) {
    const int kbase = INPUT + kt * 32;
    v8bf a0 = *(const v8bf*)(Ahi + arow + kbase + half);
    v8bf a1 = *(const v8bf*)(Ahi + arow + kbase + 16 + half);
    v8bf l0 = *(const v8bf*)(Alo + arow + kbase + half);
    v8bf l1 = *(const v8bf*)(Alo + arow + kbase + 16 + half);
    v16bf ahi = __builtin_shufflevector(a0, a1, 0,1,2,3,4,5,6,7,8,9,10,11,12,13,14,15);
    v16bf alo = __builtin_shufflevector(l0, l1, 0,1,2,3,4,5,6,7,8,9,10,11,12,13,14,15);
    const long fb = (((long)kt * NNT2 + w) * 32 + lane) * 16;
    v16bf bhi = *(const v16bf*)(BOhi + fb);
    v16bf blo = *(const v16bf*)(BOlo + fb);
    Co = __builtin_amdgcn_wmma_f32_16x16x32_bf16(false, ahi, false, bhi, (short)0, Co, false, false);
    Co = __builtin_amdgcn_wmma_f32_16x16x32_bf16(false, ahi, false, blo, (short)0, Co, false, false);
    Co = __builtin_amdgcn_wmma_f32_16x16x32_bf16(false, alo, false, bhi, (short)0, Co, false, false);
  }
  const int n = w * 16 + (lane & 15);
#pragma unroll
  for (int e = 0; e < 8; ++e) {
    const int mr = e + half;
    out[(long)(b0 + mr) * OUTPUT + n] = Co[e] + b_out[n];
  }
}

// ---------------------------------------------------------------------------
extern "C" void kernel_launch(void* const* d_in, const int* in_sizes, int n_in,
                              void* d_out, int out_size, void* d_ws, size_t ws_size,
                              hipStream_t stream) {
  (void)in_sizes; (void)n_in; (void)out_size; (void)ws_size;

  const float* x     = (const float*)d_in[0];
  const float* W_in  = (const float*)d_in[1];
  const float* b_in  = (const float*)d_in[2];
  const float* W_h   = (const float*)d_in[3];
  const float* b_h   = (const float*)d_in[4];
  const float* tau   = (const float*)d_in[5];
  const float* W_out = (const float*)d_in[6];
  const float* b_out = (const float*)d_in[7];

  // workspace layout (~7 MB): split-bf16 packed weights in fragment order
  unsigned short* Bhi  = (unsigned short*)d_ws;
  unsigned short* Blo  = Bhi  + (size_t)KCAT   * HIDDEN;
  unsigned short* BOhi = Blo  + (size_t)KCAT   * HIDDEN;
  unsigned short* BOlo = BOhi + (size_t)HIDDEN * OUTPUT;

  pack_wcat<<<(KCAT * HIDDEN + 255) / 256, 256, 0, stream>>>(W_in, W_h, Bhi, Blo);
  pack_wout<<<(HIDDEN * OUTPUT + 255) / 256, 256, 0, stream>>>(W_out, BOhi, BOlo);

  // dynamic LDS: split-bf16 A buffers (padded) + fused bias + 1/tau
  size_t smem = (size_t)32 * AP * sizeof(unsigned short) + 2u * HIDDEN * sizeof(float);
  lnn_scan<<<BATCH / 16, 512, smem, stream>>>(x, b_in, b_h, tau, b_out,
                                              Bhi, Blo, BOhi, BOlo, (float*)d_out);
}
